// RelativeCrossAttentionLayer_47536698032949
// MI455X (gfx1250) — compile-verified, matrix operand order checked by reading, not run on hardware
//
#include <hip/hip_runtime.h>
#include <hip/hip_bf16.h>

#define D_MODEL 512
#define D_INNER 1024
#define D_STATE 16
#define D_CONV  4
#define DT_RANK 32
#define BATCH   2
#define SEQ     2048
#define ROWS    (BATCH * SEQ)   // 4096

typedef __attribute__((ext_vector_type(16))) _Float16 v16h;
typedef __attribute__((ext_vector_type(8)))  _Float16 v8h;
typedef __attribute__((ext_vector_type(8)))  float    v8f;

// ---------------------------------------------------------------------------
// WMMA helper (gfx1250, wave32): D = A(16x32 f16) x B(32x16 f16) + C(16x16 f32)
// ---------------------------------------------------------------------------
__device__ __forceinline__ v8f wmma_f16(v16h a, v16h b, v8f c) {
    return __builtin_amdgcn_wmma_f32_16x16x32_f16(
        /*neg_a=*/false, a, /*neg_b=*/false, b,
        /*c_mod=*/(short)0, c, /*reuse_a=*/false, /*reuse_b=*/false);
}

// 16-bit A/B fragment layout (ISA 7.12.2): lane holds row (M or N = lane&15);
// halves 0..7 = K base..base+7, halves 8..15 = K base+16..base+23, with
// base = (lane>=16 ? 8 : 0).  Caller passes p = row_ptr + k0 + base (halves).
// Two 16-byte loads -> global_load_b128, no conversion VALU.
__device__ __forceinline__ v16h load_frag(const _Float16* __restrict__ p) {
    v8h lo = *(const v8h*)(p);
    v8h hi = *(const v8h*)(p + 16);
    return __builtin_shufflevector(lo, hi, 0, 1, 2, 3, 4, 5, 6, 7,
                                           8, 9, 10, 11, 12, 13, 14, 15);
}

__device__ __forceinline__ float siluf(float x)     { return x / (1.0f + __expf(-x)); }
__device__ __forceinline__ float softplusf_(float x){ return (x > 20.0f) ? x : log1pf(__expf(x)); }

// ---------------------------------------------------------------------------
// 0) generic f32 -> f16 staging pass (weights)
// ---------------------------------------------------------------------------
__global__ void k_cvt(const float* __restrict__ src, _Float16* __restrict__ dst, int n) {
    int id = blockIdx.x * blockDim.x + threadIdx.x;
    if (id < n) dst[id] = (_Float16)src[id];
}

// ---------------------------------------------------------------------------
// 1) ada modulation: m = silu(diff_ts) @ ada_w^T + ada_b   -> ss[b][0..1024)
// ---------------------------------------------------------------------------
__global__ void k_ada(const float* __restrict__ diff_ts, const float* __restrict__ ada_w,
                      const float* __restrict__ ada_b, float* __restrict__ ss) {
    int j = blockIdx.x * blockDim.x + threadIdx.x;       // 0..2047
    if (j >= BATCH * 2 * D_MODEL) return;
    int b = j / (2 * D_MODEL);
    int o = j % (2 * D_MODEL);
    const float* dv = diff_ts + b * D_MODEL;
    const float* w  = ada_w  + o * D_MODEL;
    float s = 0.0f;
    for (int k = 0; k < D_MODEL; ++k) s += siluf(dv[k]) * w[k];
    ss[b * 2 * D_MODEL + o] = s + ada_b[o];
}

// ---------------------------------------------------------------------------
// 2) modulation fused into f16 staging of x:  x16 = f16(q*(1+scale)+shift)
// ---------------------------------------------------------------------------
__global__ void k_mod(const float* __restrict__ q, const float* __restrict__ ss,
                      _Float16* __restrict__ x16) {
    int id  = blockIdx.x * blockDim.x + threadIdx.x;     // ROWS*D_MODEL
    int k   = id & (D_MODEL - 1);
    int row = id >> 9;
    int b   = row >> 11;
    const float* ssb = ss + b * 2 * D_MODEL;
    x16[id] = (_Float16)(q[id] * (1.0f + ssb[k]) + ssb[D_MODEL + k]);
}

// ---------------------------------------------------------------------------
// 3) in_proj GEMM: xz = x16 @ Win16^T  -> xm | z (f32)
//    One wave computes a 16(M) x 64(N) strip; K = 512 in 16 WMMA steps.
// ---------------------------------------------------------------------------
__global__ void k_inproj(const _Float16* __restrict__ x16, const _Float16* __restrict__ Win,
                         float* __restrict__ xm, float* __restrict__ z) {
    const int lane = threadIdx.x & 31;
    const int hi   = lane >> 4;
    const int gw   = blockIdx.x * (blockDim.x >> 5) + (threadIdx.x >> 5);
    const int NT64 = (2 * D_INNER) / 64;                 // 32
    const int mtile = gw / NT64;
    const int n64   = (gw % NT64) * 64;
    const int row   = mtile * 16 + (lane & 15);
    const _Float16* a_row = x16 + row * D_MODEL;
    const int wn = n64 + (lane & 15);

    v8f acc0 = {}, acc1 = {}, acc2 = {}, acc3 = {};
    for (int k0 = 0; k0 < D_MODEL; k0 += 32) {
        const int kb = k0 + hi * 8;
        v16h a  = load_frag(a_row + kb);
        v16h b0 = load_frag(Win + (wn +  0) * D_MODEL + kb);
        v16h b1 = load_frag(Win + (wn + 16) * D_MODEL + kb);
        v16h b2 = load_frag(Win + (wn + 32) * D_MODEL + kb);
        v16h b3 = load_frag(Win + (wn + 48) * D_MODEL + kb);
        if (k0 + 32 < D_MODEL)
            __builtin_prefetch(Win + wn * D_MODEL + kb + 32, 0, 3);
        acc0 = wmma_f16(a, b0, acc0);
        acc1 = wmma_f16(a, b1, acc1);
        acc2 = wmma_f16(a, b2, acc2);
        acc3 = wmma_f16(a, b3, acc3);
    }
    float* dst = (n64 < D_INNER) ? xm : z;
    const int coff = (n64 < D_INNER) ? n64 : (n64 - D_INNER);
    const int c = lane & 15;
    v8f accs[4] = {acc0, acc1, acc2, acc3};
#pragma unroll
    for (int s = 0; s < 4; ++s) {
#pragma unroll
        for (int r = 0; r < 8; ++r) {
            int ro = mtile * 16 + r + 8 * hi;
            dst[ro * D_INNER + coff + s * 16 + c] = accs[s][r];
        }
    }
}

// ---------------------------------------------------------------------------
// 4) depthwise causal conv (width 4) + bias + SiLU -> xc (f32) and xc16 (f16)
// ---------------------------------------------------------------------------
__global__ void k_conv(const float* __restrict__ xm, const float* __restrict__ conv_w,
                       const float* __restrict__ conv_b,
                       float* __restrict__ xc, _Float16* __restrict__ xc16) {
    int id = blockIdx.x * blockDim.x + threadIdx.x;
    if (id >= ROWS * D_INNER) return;
    int d   = id & (D_INNER - 1);
    int row = id >> 10;
    int l   = row & (SEQ - 1);
    float s = conv_b[d];
#pragma unroll
    for (int k = 0; k < D_CONV; ++k) {
        int ls = l - (D_CONV - 1) + k;
        if (ls >= 0) s += xm[(row - (D_CONV - 1) + k) * D_INNER + d] * conv_w[d * D_CONV + k];
    }
    float v = siluf(s);
    xc[id]   = v;
    xc16[id] = (_Float16)v;
}

// ---------------------------------------------------------------------------
// 5) x_proj: xdb = xc16 @ Wx16^T (N = 64) -> dt16 (f16) | Bm | Cm (f32)
// ---------------------------------------------------------------------------
__global__ void k_xdb(const _Float16* __restrict__ xc16, const _Float16* __restrict__ Wx,
                      _Float16* __restrict__ dt16, float* __restrict__ Bm,
                      float* __restrict__ Cm) {
    const int lane = threadIdx.x & 31;
    const int hi   = lane >> 4;
    const int mtile = blockIdx.x * (blockDim.x >> 5) + (threadIdx.x >> 5);
    const int row   = mtile * 16 + (lane & 15);
    const int wn    = lane & 15;

    v8f a0 = {}, a1 = {}, a2 = {}, a3 = {};
    for (int k0 = 0; k0 < D_INNER; k0 += 32) {
        const int kb = k0 + hi * 8;
        v16h a  = load_frag(xc16 + row * D_INNER + kb);
        v16h b0 = load_frag(Wx + (wn +  0) * D_INNER + kb);
        v16h b1 = load_frag(Wx + (wn + 16) * D_INNER + kb);
        v16h b2 = load_frag(Wx + (wn + 32) * D_INNER + kb);
        v16h b3 = load_frag(Wx + (wn + 48) * D_INNER + kb);
        a0 = wmma_f16(a, b0, a0);
        a1 = wmma_f16(a, b1, a1);
        a2 = wmma_f16(a, b2, a2);
        a3 = wmma_f16(a, b3, a3);
    }
    const int c = lane & 15;
#pragma unroll
    for (int r = 0; r < 8; ++r) {
        int ro = mtile * 16 + r + 8 * hi;
        dt16[ro * DT_RANK + c]      = (_Float16)a0[r];
        dt16[ro * DT_RANK + 16 + c] = (_Float16)a1[r];
        Bm[ro * D_STATE + c]        = a2[r];
        Cm[ro * D_STATE + c]        = a3[r];
    }
}

// ---------------------------------------------------------------------------
// 6) dt_proj + softplus: delta = softplus(dt16 @ Wdt16^T + bdt)  (K=32: 1 WMMA)
// ---------------------------------------------------------------------------
__global__ void k_delta(const _Float16* __restrict__ dt16, const _Float16* __restrict__ Wdt,
                        const float* __restrict__ bdt, float* __restrict__ delta) {
    const int lane = threadIdx.x & 31;
    const int hi   = lane >> 4;
    const int gw   = blockIdx.x * (blockDim.x >> 5) + (threadIdx.x >> 5);
    const int NT64 = D_INNER / 64;                       // 16
    const int mtile = gw / NT64;
    const int n64   = (gw % NT64) * 64;
    const int row   = mtile * 16 + (lane & 15);
    const int kb    = hi * 8;

    v16h a = load_frag(dt16 + row * DT_RANK + kb);
    const int c = lane & 15;
#pragma unroll
    for (int s = 0; s < 4; ++s) {
        int n = n64 + s * 16 + c;
        v16h bf = load_frag(Wdt + n * DT_RANK + kb);
        v8f zc = {};
        v8f acc = wmma_f16(a, bf, zc);
#pragma unroll
        for (int r = 0; r < 8; ++r) {
            int ro = mtile * 16 + r + 8 * hi;
            delta[ro * D_INNER + n] = softplusf_(acc[r] + bdt[n]);
        }
    }
}

// ---------------------------------------------------------------------------
// 7) selective scan, fused with  y = (scan + xc*D) * silu(z) -> y16 (f16).
//    One lane per (b,d) channel, h[16] in registers, 2048 sequential steps.
// ---------------------------------------------------------------------------
__global__ void k_scan(const float* __restrict__ delta, const float* __restrict__ xcbuf,
                       const float* __restrict__ Bm, const float* __restrict__ Cm,
                       const float* __restrict__ zbuf, const float* __restrict__ A_log,
                       const float* __restrict__ D_param, _Float16* __restrict__ y16) {
    int gtid = blockIdx.x * blockDim.x + threadIdx.x;    // 0..2047
    int b = gtid >> 10;
    int d = gtid & (D_INNER - 1);

    float A[D_STATE];
#pragma unroll
    for (int n = 0; n < D_STATE; ++n) A[n] = -__expf(A_log[d * D_STATE + n]);
    const float Dp = D_param[d];

    float h[D_STATE];
#pragma unroll
    for (int n = 0; n < D_STATE; ++n) h[n] = 0.0f;

    const int base = b * SEQ;
    for (int t = 0; t < SEQ; ++t) {
        const int ri = base + t;
        float dlt = delta[ri * D_INNER + d];
        float xt  = xcbuf[ri * D_INNER + d];
        float zt  = zbuf [ri * D_INNER + d];
        float du  = dlt * xt;
        float y   = 0.0f;
        const float4* Bv = (const float4*)(Bm + ri * D_STATE);  // block-uniform
        const float4* Cv = (const float4*)(Cm + ri * D_STATE);
#pragma unroll
        for (int q4 = 0; q4 < 4; ++q4) {
            float4 Bq = Bv[q4], Cq = Cv[q4];
            float bb[4] = {Bq.x, Bq.y, Bq.z, Bq.w};
            float cc[4] = {Cq.x, Cq.y, Cq.z, Cq.w};
#pragma unroll
            for (int j = 0; j < 4; ++j) {
                int n = q4 * 4 + j;
                float dA = __expf(dlt * A[n]);
                h[n] = dA * h[n] + du * bb[j];
                y += h[n] * cc[j];
            }
        }
        y16[ri * D_INNER + d] = (_Float16)((y + xt * Dp) * siluf(zt));
    }
}

// ---------------------------------------------------------------------------
// 8) out_proj: attn = y16 @ Wo16^T  (4096 x 512, K = 1024)
// ---------------------------------------------------------------------------
__global__ void k_outproj(const _Float16* __restrict__ y16, const _Float16* __restrict__ Wo,
                          float* __restrict__ attn) {
    const int lane = threadIdx.x & 31;
    const int hi   = lane >> 4;
    const int gw   = blockIdx.x * (blockDim.x >> 5) + (threadIdx.x >> 5);
    const int NT64 = D_MODEL / 64;                       // 8
    const int mtile = gw / NT64;
    const int n64   = (gw % NT64) * 64;
    const int row   = mtile * 16 + (lane & 15);
    const int wn    = n64 + (lane & 15);

    v8f acc0 = {}, acc1 = {}, acc2 = {}, acc3 = {};
    for (int k0 = 0; k0 < D_INNER; k0 += 32) {
        const int kb = k0 + hi * 8;
        v16h a  = load_frag(y16 + row * D_INNER + kb);
        v16h b0 = load_frag(Wo + (wn +  0) * D_INNER + kb);
        v16h b1 = load_frag(Wo + (wn + 16) * D_INNER + kb);
        v16h b2 = load_frag(Wo + (wn + 32) * D_INNER + kb);
        v16h b3 = load_frag(Wo + (wn + 48) * D_INNER + kb);
        acc0 = wmma_f16(a, b0, acc0);
        acc1 = wmma_f16(a, b1, acc1);
        acc2 = wmma_f16(a, b2, acc2);
        acc3 = wmma_f16(a, b3, acc3);
    }
    const int c = lane & 15;
    v8f accs[4] = {acc0, acc1, acc2, acc3};
#pragma unroll
    for (int s = 0; s < 4; ++s) {
#pragma unroll
        for (int r = 0; r < 8; ++r) {
            int ro = mtile * 16 + r + 8 * hi;
            attn[ro * D_MODEL + n64 + s * 16 + c] = accs[s][r];
        }
    }
}

// ---------------------------------------------------------------------------
// 9) residual + LayerNorm: out = LN(query + attn) * g + b   (one block / row)
// ---------------------------------------------------------------------------
__global__ void k_ln(const float* __restrict__ query, const float* __restrict__ attn,
                     const float* __restrict__ g, const float* __restrict__ bt,
                     float* __restrict__ out) {
    const int row = blockIdx.x;
    const int tid = threadIdx.x;                         // 256 threads
    float v0 = query[row * D_MODEL + tid]       + attn[row * D_MODEL + tid];
    float v1 = query[row * D_MODEL + tid + 256] + attn[row * D_MODEL + tid + 256];
    __shared__ float ssum[256], ssq[256];
    ssum[tid] = v0 + v1;
    ssq[tid]  = v0 * v0 + v1 * v1;
    __syncthreads();
    for (int off = 128; off > 0; off >>= 1) {
        if (tid < off) { ssum[tid] += ssum[tid + off]; ssq[tid] += ssq[tid + off]; }
        __syncthreads();
    }
    float mu   = ssum[0] * (1.0f / D_MODEL);
    float var  = ssq[0] * (1.0f / D_MODEL) - mu * mu;
    float rstd = rsqrtf(var + 1e-5f);
    out[row * D_MODEL + tid]       = (v0 - mu) * rstd * g[tid]       + bt[tid];
    out[row * D_MODEL + tid + 256] = (v1 - mu) * rstd * g[tid + 256] + bt[tid + 256];
}

// ---------------------------------------------------------------------------
extern "C" void kernel_launch(void* const* d_in, const int* in_sizes, int n_in,
                              void* d_out, int out_size, void* d_ws, size_t ws_size,
                              hipStream_t stream) {
    (void)in_sizes; (void)n_in; (void)out_size; (void)ws_size;
    const float* query     = (const float*)d_in[0];
    // d_in[1] (value) is unused by the reference
    const float* diff_ts   = (const float*)d_in[2];
    const float* ada_w     = (const float*)d_in[3];
    const float* ada_b     = (const float*)d_in[4];
    const float* in_proj_w = (const float*)d_in[5];
    const float* conv_w    = (const float*)d_in[6];
    const float* conv_b    = (const float*)d_in[7];
    const float* x_proj_w  = (const float*)d_in[8];
    const float* dt_proj_w = (const float*)d_in[9];
    const float* dt_proj_b = (const float*)d_in[10];
    const float* A_log     = (const float*)d_in[11];
    const float* D_param   = (const float*)d_in[12];
    const float* out_proj_w= (const float*)d_in[13];
    const float* ln_g      = (const float*)d_in[14];
    const float* ln_b      = (const float*)d_in[15];
    float* out = (float*)d_out;

    // ---- workspace layout (bytes); delta reuses xm, attn reuses z ----
    char* w = (char*)d_ws;
    float* ss   = (float*)w;  w += 4096 * sizeof(float);
    float* xm   = (float*)w;  w += (size_t)ROWS * D_INNER * sizeof(float);
    float* z    = (float*)w;  w += (size_t)ROWS * D_INNER * sizeof(float);
    float* xc   = (float*)w;  w += (size_t)ROWS * D_INNER * sizeof(float);
    float* Bm   = (float*)w;  w += (size_t)ROWS * D_STATE * sizeof(float);
    float* Cm   = (float*)w;  w += (size_t)ROWS * D_STATE * sizeof(float);
    _Float16* x16   = (_Float16*)w; w += (size_t)ROWS * D_MODEL * sizeof(_Float16);
    _Float16* Win16 = (_Float16*)w; w += (size_t)2 * D_INNER * D_MODEL * sizeof(_Float16);
    _Float16* Wx16  = (_Float16*)w; w += (size_t)(DT_RANK + 2 * D_STATE) * D_INNER * sizeof(_Float16);
    _Float16* Wdt16 = (_Float16*)w; w += (size_t)D_INNER * DT_RANK * sizeof(_Float16);
    _Float16* Wo16  = (_Float16*)w; w += (size_t)D_MODEL * D_INNER * sizeof(_Float16);
    _Float16* xc16  = (_Float16*)w; w += (size_t)ROWS * D_INNER * sizeof(_Float16);
    _Float16* dt16  = (_Float16*)w; w += (size_t)ROWS * DT_RANK * sizeof(_Float16);
    _Float16* y16   = (_Float16*)w; w += (size_t)ROWS * D_INNER * sizeof(_Float16);
    float* delta = xm;    // xm dead after conv
    float* attn  = z;     // z dead after scan

    // ---- weight staging (one pass each) ----
    k_cvt<<<(2 * D_INNER * D_MODEL) / 256, 256, 0, stream>>>(in_proj_w, Win16, 2 * D_INNER * D_MODEL);
    k_cvt<<<((DT_RANK + 2 * D_STATE) * D_INNER) / 256, 256, 0, stream>>>(x_proj_w, Wx16, (DT_RANK + 2 * D_STATE) * D_INNER);
    k_cvt<<<(D_INNER * DT_RANK) / 256, 256, 0, stream>>>(dt_proj_w, Wdt16, D_INNER * DT_RANK);
    k_cvt<<<(D_MODEL * D_INNER) / 256, 256, 0, stream>>>(out_proj_w, Wo16, D_MODEL * D_INNER);

    // ---- pipeline ----
    k_ada    <<<8,    256, 0, stream>>>(diff_ts, ada_w, ada_b, ss);
    k_mod    <<<(ROWS * D_MODEL) / 256, 256, 0, stream>>>(query, ss, x16);
    k_inproj <<<1024, 256, 0, stream>>>(x16, Win16, xm, z);                 // 8192 wave-jobs
    k_conv   <<<(ROWS * D_INNER) / 256, 256, 0, stream>>>(xm, conv_w, conv_b, xc, xc16);
    k_xdb    <<<32,   256, 0, stream>>>(xc16, Wx16, dt16, Bm, Cm);          // 256 wave-jobs
    k_delta  <<<512,  256, 0, stream>>>(dt16, Wdt16, dt_proj_b, delta);     // 4096 wave-jobs
    k_scan   <<<8,    256, 0, stream>>>(delta, xc, Bm, Cm, z, A_log, D_param, y16);
    k_outproj<<<256,  256, 0, stream>>>(y16, Wo16, attn);                   // 2048 wave-jobs
    k_ln     <<<ROWS, 256, 0, stream>>>(query, attn, ln_g, ln_b, out);
}
